// GATEncoder_61057255080454
// MI455X (gfx1250) — matrix operand order repaired and checked by reading, not compiled, not run on hardware
//
#include <hip/hip_runtime.h>
#include <math.h>

// ---------------- problem constants (from reference) ----------------
#define BATCH   64
#define NMAXQ   299
#define NN      300
#define TSUB    8
#define DD      512
#define HH      512
#define NHEADS  8
#define NHID    64
#define NEGV    (-9.0e15f)
#define SCLD    320       // attention matrix leading dim, padded to mult of 32
#define SCPAD   (SCLD - NN)

typedef __attribute__((ext_vector_type(16))) _Float16 v16h;
typedef __attribute__((ext_vector_type(8)))  float    v8f;
typedef __attribute__((ext_vector_type(4)))  float    v4f;

// =====================================================================
// Generic WMMA GEMM:  C[M,Nc] = A[M,K] @ B (+bias), optional B^T,
// batched over blockIdx.z. wave32; block = 128 threads = 4 waves; each
// wave computes a 16x64 strip (one A fragment reused across 4 B
// fragments -> 4 v_wmma per K-step of 32). f32->f16 convert on load,
// f32 accumulate. Fragment k order: contiguous, k = 16*(lane/16) + s.
//
// Hot loop is branchless: K is always a multiple of 32 (data-level
// padding with zero columns where needed), and M/N tile overreads are
// legal-by-construction (they only affect unstored C rows/cols).
// A fragment: 16 contiguous floats per lane -> 4x b128 loads.
// B^T fragment: 16 contiguous floats per lane   -> 4x b128 loads.
// B   fragment: 16 lane-coalesced strided loads.
// =====================================================================
__device__ __forceinline__ v16h load_a16(const float* __restrict__ p) {
  const v4f x0 = *(const v4f*)(p);
  const v4f x1 = *(const v4f*)(p + 4);
  const v4f x2 = *(const v4f*)(p + 8);
  const v4f x3 = *(const v4f*)(p + 12);
  v16h f;
  f[0]  = (_Float16)x0.x; f[1]  = (_Float16)x0.y; f[2]  = (_Float16)x0.z; f[3]  = (_Float16)x0.w;
  f[4]  = (_Float16)x1.x; f[5]  = (_Float16)x1.y; f[6]  = (_Float16)x1.z; f[7]  = (_Float16)x1.w;
  f[8]  = (_Float16)x2.x; f[9]  = (_Float16)x2.y; f[10] = (_Float16)x2.z; f[11] = (_Float16)x2.w;
  f[12] = (_Float16)x3.x; f[13] = (_Float16)x3.y; f[14] = (_Float16)x3.z; f[15] = (_Float16)x3.w;
  return f;
}

__device__ __forceinline__ v16h load_b16_contig(const float* __restrict__ p) {
  const v4f x0 = *(const v4f*)(p);
  const v4f x1 = *(const v4f*)(p + 4);
  const v4f x2 = *(const v4f*)(p + 8);
  const v4f x3 = *(const v4f*)(p + 12);
  v16h f;
  f[0]  = (_Float16)x0.x; f[1]  = (_Float16)x0.y; f[2]  = (_Float16)x0.z; f[3]  = (_Float16)x0.w;
  f[4]  = (_Float16)x1.x; f[5]  = (_Float16)x1.y; f[6]  = (_Float16)x1.z; f[7]  = (_Float16)x1.w;
  f[8]  = (_Float16)x2.x; f[9]  = (_Float16)x2.y; f[10] = (_Float16)x2.z; f[11] = (_Float16)x2.w;
  f[12] = (_Float16)x3.x; f[13] = (_Float16)x3.y; f[14] = (_Float16)x3.z; f[15] = (_Float16)x3.w;
  return f;
}

__device__ __forceinline__ v16h load_b16_strided(const float* __restrict__ p, int ldb) {
  v16h f;
#pragma unroll
  for (int h = 0; h < 16; ++h) f[h] = (_Float16)p[(long)h * ldb];
  return f;
}

template <int TRANSB>
__global__ __launch_bounds__(128) void gemm_wmma_kernel(
    const float* __restrict__ A, long long sAz, int lda,
    const float* __restrict__ Bm, long long sBz, int ldb,
    float* __restrict__ C, long long sCz, int ldc,
    const float* __restrict__ bias,
    int M, int K, int Nc) {
  const int z = blockIdx.z;
  A  += (long long)z * sAz;
  Bm += (long long)z * sBz;
  C  += (long long)z * sCz;

  const int lane = threadIdx.x & 31;
  const int wid  = threadIdx.x >> 5;            // 0..3
  const int row0 = blockIdx.y * 64 + wid * 16;
  const int col0 = blockIdx.x * 64;

  // A: lane holds row row0+(lane&15); 16 contiguous k values at k+0
  //    (+16 for hi lanes).
  const float* ap = A + (long)(row0 + (lane & 15)) * lda + ((lane >> 4) << 4);

  v8f acc[4] = {v8f{}, v8f{}, v8f{}, v8f{}};

  if (TRANSB) {
    // B^T: lane holds col col0+nb*16+(lane&15); 16 contiguous k values.
    const float* bp = Bm + (long)(col0 + (lane & 15)) * ldb + ((lane >> 4) << 4);
    for (int k0 = 0; k0 < K; k0 += 32) {
      const v16h a = load_a16(ap + k0);
#pragma unroll
      for (int nb = 0; nb < 4; ++nb) {
        const v16h b = load_b16_contig(bp + (long)(nb * 16) * ldb + k0);
        acc[nb] = __builtin_amdgcn_wmma_f32_16x16x32_f16(
            false, a, false, b, (short)0, acc[nb], false, false);
      }
    }
  } else {
    // B: element (k, c); per-half stride ldb, coalesced across lanes.
    const float* bp = Bm + (long)((lane >> 4) << 4) * ldb + col0 + (lane & 15);
    for (int k0 = 0; k0 < K; k0 += 32) {
      const v16h a = load_a16(ap + k0);
      const float* bk = bp + (long)k0 * ldb;
#pragma unroll
      for (int nb = 0; nb < 4; ++nb) {
        const v16h b = load_b16_strided(bk + nb * 16, ldb);
        acc[nb] = __builtin_amdgcn_wmma_f32_16x16x32_f16(
            false, a, false, b, (short)0, acc[nb], false, false);
      }
    }
  }

  // store (cold; bounds-checked)
  const int cn   = lane & 15;
  const int roff = (lane >> 4) << 3;
#pragma unroll
  for (int nb = 0; nb < 4; ++nb) {
    const int c = col0 + nb * 16 + cn;
    if (c >= Nc) continue;
    const float bv = bias ? bias[c] : 0.0f;
#pragma unroll
    for (int v = 0; v < 8; ++v) {
      const int r = row0 + roff + v;
      if (r < M) C[(long)r * ldc + c] = acc[nb][v] + bv;
    }
  }
}

// =====================================================================
// Elementwise / reduction kernels
// =====================================================================

__global__ void embed_kernel(const int* __restrict__ tokens,
                             const int* __restrict__ tok_lens,
                             const int* __restrict__ seq_lens,
                             const float* __restrict__ emb,
                             float* __restrict__ outp) {
  const int bn = blockIdx.x;
  const int b = bn / NN, n = bn % NN;
  const int nc = seq_lens[b] - 1;
  for (int c = threadIdx.x; c < DD; c += blockDim.x) {
    float val;
    if (n < nc) {
      float s = 0.f;
      const int tl = tok_lens[b * NMAXQ + n];
      for (int j = 0; j < tl; ++j) {
        const int tok = tokens[(b * NMAXQ + n) * TSUB + j];
        s += emb[(long)tok * DD + c];
      }
      val = s / (float)nc;
    } else if (n == nc) {
      val = emb[2 * DD + c];        // eos
    } else {
      val = emb[c];                 // padding
    }
    outp[((long)b * NN + n) * DD + c] = val;
  }
}

// zero the padded K columns (m in [NN, SCLD)) of the attention buffer
__global__ void zero_scpad_kernel(float* __restrict__ s) {
  const int i = blockIdx.x * blockDim.x + threadIdx.x;
  if (i >= BATCH * NN * SCPAD) return;
  const int m  = NN + (i % SCPAD);
  const int bn = i / SCPAD;
  s[(long)bn * SCLD + m] = 0.f;
}

// softmax over the BATCH axis of scores (B,N,N) stored with ld=SCLD, in place
__global__ void softmax_axis0_kernel(float* __restrict__ s) {
  const int idx = blockIdx.x * blockDim.x + threadIdx.x;
  if (idx >= NN * NN) return;
  const int n = idx / NN, m = idx % NN;
  const long off = (long)n * SCLD + m;
  const long bs = (long)NN * SCLD;
  float mx = -3.4e38f;
  for (int b = 0; b < BATCH; ++b) mx = fmaxf(mx, s[b * bs + off]);
  float sum = 0.f;
  for (int b = 0; b < BATCH; ++b) {
    const float e = expf(s[b * bs + off] - mx);
    s[b * bs + off] = e;
    sum += e;
  }
  const float inv = 1.f / sum;
  for (int b = 0; b < BATCH; ++b) s[b * bs + off] *= inv;
}

// att = softmax(where(cond, s, NEG), axis=1) -- softmax over n per (b,m)
__global__ void mask_softmax_axis1_kernel(float* __restrict__ s,
                                          const float* __restrict__ adj) {
  const int idx = blockIdx.x * blockDim.x + threadIdx.x;  // b*NN + m
  if (idx >= BATCH * NN) return;
  const int b = idx / NN, m = idx % NN;
  const long sbase = (long)b * NN * SCLD;
  const long abase = (long)b * NN * NN;
  float mx = -3.4e38f;
  for (int n = 0; n < NN; ++n) {
    const bool c = (adj[abase + (long)n * NN + m] + adj[abase + (long)m * NN + n] +
                    (n == m ? 1.f : 0.f)) > 0.f;
    mx = fmaxf(mx, c ? s[sbase + (long)n * SCLD + m] : NEGV);
  }
  float sum = 0.f;
  for (int n = 0; n < NN; ++n) {
    const bool c = (adj[abase + (long)n * NN + m] + adj[abase + (long)m * NN + n] +
                    (n == m ? 1.f : 0.f)) > 0.f;
    sum += expf((c ? s[sbase + (long)n * SCLD + m] : NEGV) - mx);
  }
  const float inv = 1.f / sum;
  for (int n = 0; n < NN; ++n) {
    const bool c = (adj[abase + (long)n * NN + m] + adj[abase + (long)m * NN + n] +
                    (n == m ? 1.f : 0.f)) > 0.f;
    s[sbase + (long)n * SCLD + m] =
        expf((c ? s[sbase + (long)n * SCLD + m] : NEGV) - mx) * inv;
  }
}

// softmax over node axis n for each (b,f); optional trailing ELU
__global__ void softmax_nodes_kernel(float* __restrict__ x, int ld, int F, int do_elu) {
  const int idx = blockIdx.x * blockDim.x + threadIdx.x;  // b*F + f
  if (idx >= BATCH * F) return;
  const int b = idx / F, f = idx % F;
  float* p = x + (long)b * NN * ld + f;
  float mx = -3.4e38f;
  for (int n = 0; n < NN; ++n) mx = fmaxf(mx, p[(long)n * ld]);
  float sum = 0.f;
  for (int n = 0; n < NN; ++n) sum += expf(p[(long)n * ld] - mx);
  const float inv = 1.f / sum;
  for (int n = 0; n < NN; ++n) {
    float v = expf(p[(long)n * ld] - mx) * inv;
    if (do_elu) v = (v > 0.f) ? v : (expf(v) - 1.f);
    p[(long)n * ld] = v;
  }
}

// xt_f[t,b,:] = out2[b,t,:] ; xt_b[s,b,:] = out2[b, clip(L-1-s,0), :]
__global__ void make_xt_kernel(const float* __restrict__ out2,
                               const int* __restrict__ seq_lens,
                               float* __restrict__ xtf, float* __restrict__ xtb) {
  const long i = (long)blockIdx.x * blockDim.x + threadIdx.x;
  if (i >= (long)NN * BATCH * HH) return;
  const int  c  = (int)(i % HH);
  const long tb = i / HH;
  const int  b  = (int)(tb % BATCH);
  const int  t  = (int)(tb / BATCH);
  xtf[i] = out2[((long)b * NN + t) * HH + c];
  int id = seq_lens[b] - 1 - t;
  if (id < 0) id = 0;
  xtb[i] = out2[((long)b * NN + id) * HH + c];
}

__global__ void zero_kernel(float* __restrict__ p, int n) {
  const int i = blockIdx.x * blockDim.x + threadIdx.x;
  if (i < n) p[i] = 0.f;
}

// one GRU step: combine precomputed gi with fresh gh, update h, emit output
__global__ void gru_gates_kernel(const float* __restrict__ gi,
                                 const float* __restrict__ gh,
                                 float* __restrict__ h,
                                 const int* __restrict__ seq_lens,
                                 float* __restrict__ outp, int t, int dir) {
  const int i = blockIdx.x * blockDim.x + threadIdx.x;
  if (i >= BATCH * HH) return;
  const int b = i / HH, c = i % HH;
  const int L = seq_lens[b];
  const bool m = (t < L);
  const int g = b * 3 * HH + c;
  const float r  = 1.f / (1.f + expf(-(gi[g]          + gh[g])));
  const float z  = 1.f / (1.f + expf(-(gi[g + HH]     + gh[g + HH])));
  const float nn = tanhf(gi[g + 2 * HH] + r * gh[g + 2 * HH]);
  const float hnew = (1.f - z) * nn + z * h[i];
  if (dir == 0) {
    outp[((long)t * BATCH + b) * HH + c] = m ? hnew : 0.f;       // out_f
  } else if (m) {
    outp[((long)(L - 1 - t) * BATCH + b) * HH + c] += hnew;      // out_b scatter
  }
  if (m) h[i] = hnew;
}

__global__ void copy_hidden_kernel(const float* __restrict__ hF,
                                   const float* __restrict__ hB,
                                   float* __restrict__ outp) {
  const int i = blockIdx.x * blockDim.x + threadIdx.x;
  if (i >= BATCH * HH) return;
  const long off = (long)NN * BATCH * HH;
  outp[off + i] = hF[i];
  outp[off + BATCH * HH + i] = hB[i];
}

// =====================================================================
// Host-side orchestration
// =====================================================================
extern "C" void kernel_launch(void* const* d_in, const int* in_sizes, int n_in,
                              void* d_out, int out_size, void* d_ws, size_t ws_size,
                              hipStream_t stream) {
  (void)in_sizes; (void)n_in; (void)out_size; (void)ws_size;

  const int*   tokens   = (const int*)d_in[0];
  const int*   tok_lens = (const int*)d_in[1];
  const int*   seq_lens = (const int*)d_in[2];
  const float* adjacent = (const float*)d_in[3];
  const float* emb      = (const float*)d_in[4];
  const float* Wq       = (const float*)d_in[5];
  const float* Wk       = (const float*)d_in[6];
  const float* Wq_o     = (const float*)d_in[7];
  const float* Wk_o     = (const float*)d_in[8];
  const float* gWih     = (const float*)d_in[9];
  const float* gWhh     = (const float*)d_in[10];
  const float* gbih     = (const float*)d_in[11];
  const float* gbhh     = (const float*)d_in[12];
  float* out = (float*)d_out;
  float* ws  = (float*)d_ws;

  // ---- workspace layout (floats); region reused across phases ----
  const long long SZ_BND = (long long)BATCH * NN * DD;      // 9,830,400
  const long long SZ_SC  = (long long)BATCH * NN * SCLD;    // 6,144,000
  const long long SZ_GI  = (long long)NN * BATCH * 3 * HH;  // 29,491,200
  float* embedded = ws;
  float* xcat     = ws + SZ_BND;
  float* out2     = ws + 2 * SZ_BND;
  float* region   = ws + 3 * SZ_BND;
  // phase 1 (GAT) aliases
  float* qb = region;
  float* kb = region + SZ_BND;
  float* sc = region + 2 * SZ_BND;
  float* hp = region + 2 * SZ_BND + SZ_SC;
  // phase 2 (GRU) aliases (overlap phase 1; phase-1 data dead by then)
  float* xtf = region;
  float* xtb = region + SZ_BND;
  float* gif = region + 2 * SZ_BND;
  float* gib = gif + SZ_GI;
  float* smallb = gib + SZ_GI;
  float* hF = smallb;
  float* hB = smallb + BATCH * HH;
  float* gh = smallb + 2 * BATCH * HH;

  auto gemm = [&](const float* A, long long sAz, int lda,
                  const float* Bm, long long sBz, int ldb, int transB,
                  float* C, long long sCz, int ldc, const float* bias,
                  int M, int K, int Nc, int Z) {
    dim3 g((Nc + 63) / 64, (M + 63) / 64, Z);
    if (transB)
      gemm_wmma_kernel<1><<<g, 128, 0, stream>>>(A, sAz, lda, Bm, sBz, ldb,
                                                 C, sCz, ldc, bias, M, K, Nc);
    else
      gemm_wmma_kernel<0><<<g, 128, 0, stream>>>(A, sAz, lda, Bm, sBz, ldb,
                                                 C, sCz, ldc, bias, M, K, Nc);
  };

  // 1) node embeddings
  embed_kernel<<<BATCH * NN, 256, 0, stream>>>(tokens, tok_lens, seq_lens, emb, embedded);

  // zero the K-padding columns of the attention buffer once (never written
  // afterwards inside phase 1; makes the att@x GEMM K=SCLD branchless)
  zero_scpad_kernel<<<(BATCH * NN * SCPAD + 255) / 256, 256, 0, stream>>>(sc);

  // 2) GAT-1 projections, all heads batched over blockIdx.z
  gemm(embedded, 0, DD, Wq, (long long)DD * NHID, NHID, 0,
       qb, NHID, DD, nullptr, BATCH * NN, DD, NHID, NHEADS);
  gemm(embedded, 0, DD, Wk, (long long)DD * NHID, NHID, 0,
       kb, NHID, DD, nullptr, BATCH * NN, DD, NHID, NHEADS);

  // 3) per-head attention
  for (int h = 0; h < NHEADS; ++h) {
    gemm(qb + h * NHID, (long long)NN * DD, DD,
         kb + h * NHID, (long long)NN * DD, DD, 1,
         sc, SZ_SC / BATCH, SCLD, nullptr, NN, NHID, NN, BATCH);
    softmax_axis0_kernel<<<(NN * NN + 255) / 256, 256, 0, stream>>>(sc);
    mask_softmax_axis1_kernel<<<(BATCH * NN + 255) / 256, 256, 0, stream>>>(sc, adjacent);
    gemm(sc, SZ_SC / BATCH, SCLD, embedded, (long long)NN * DD, DD, 0,
         hp, (long long)NN * DD, DD, nullptr, NN, SCLD, DD, BATCH);
    gemm(hp, 0, DD, Wk + (long long)h * DD * NHID, 0, NHID, 0,
         xcat + h * NHID, 0, DD, nullptr, BATCH * NN, DD, NHID, 1);
    softmax_nodes_kernel<<<(BATCH * NHID + 255) / 256, 256, 0, stream>>>(
        xcat + h * NHID, DD, NHID, 0);
  }

  // 4) GAT-2 (output layer) + ELU
  gemm(xcat, 0, DD, Wq_o, 0, HH, 0, qb, 0, HH, nullptr, BATCH * NN, DD, HH, 1);
  gemm(xcat, 0, DD, Wk_o, 0, HH, 0, kb, 0, HH, nullptr, BATCH * NN, DD, HH, 1);
  gemm(qb, (long long)NN * HH, HH, kb, (long long)NN * HH, HH, 1,
       sc, SZ_SC / BATCH, SCLD, nullptr, NN, HH, NN, BATCH);
  softmax_axis0_kernel<<<(NN * NN + 255) / 256, 256, 0, stream>>>(sc);
  mask_softmax_axis1_kernel<<<(BATCH * NN + 255) / 256, 256, 0, stream>>>(sc, adjacent);
  gemm(sc, SZ_SC / BATCH, SCLD, xcat, (long long)NN * DD, DD, 0,
       hp, (long long)NN * DD, DD, nullptr, NN, SCLD, DD, BATCH);
  gemm(hp, 0, DD, Wk_o, 0, HH, 0, out2, 0, HH, nullptr, BATCH * NN, DD, HH, 1);
  softmax_nodes_kernel<<<(BATCH * HH + 255) / 256, 256, 0, stream>>>(out2, HH, HH, 1);

  // 5) GRU: time-major inputs (forward + time-reversed gather)
  make_xt_kernel<<<(int)((SZ_BND + 255) / 256), 256, 0, stream>>>(out2, seq_lens, xtf, xtb);

  // 6) precompute ALL input projections gi = x_t @ Wih^T + bih (both dirs)
  gemm(xtf, 0, HH, gWih, 0, DD, 1, gif, 0, 3 * HH, gbih,
       NN * BATCH, HH, 3 * HH, 1);
  gemm(xtb, 0, HH, gWih + 3 * HH * DD, 0, DD, 1, gib, 0, 3 * HH, gbih + 3 * HH,
       NN * BATCH, HH, 3 * HH, 1);

  // 7) init hidden states
  zero_kernel<<<(2 * BATCH * HH + 255) / 256, 256, 0, stream>>>(hF, 2 * BATCH * HH);

  // 8) sequential recurrences (only h@Whh^T per step remains serialized)
  for (int t = 0; t < NN; ++t) {
    gemm(hF, 0, HH, gWhh, 0, HH, 1, gh, 0, 3 * HH, gbhh, BATCH, HH, 3 * HH, 1);
    gru_gates_kernel<<<(BATCH * HH + 255) / 256, 256, 0, stream>>>(
        gif + (long long)t * BATCH * 3 * HH, gh, hF, seq_lens, out, t, 0);
  }
  for (int t = 0; t < NN; ++t) {
    gemm(hB, 0, HH, gWhh + 3 * HH * HH, 0, HH, 1, gh, 0, 3 * HH, gbhh + 3 * HH,
         BATCH, HH, 3 * HH, 1);
    gru_gates_kernel<<<(BATCH * HH + 255) / 256, 256, 0, stream>>>(
        gib + (long long)t * BATCH * 3 * HH, gh, hB, seq_lens, out, t, 1);
  }

  // 9) final hidden states
  copy_hidden_kernel<<<(BATCH * HH + 255) / 256, 256, 0, stream>>>(hF, hB, out);
}
